// GAT_struc_87479893885521
// MI455X (gfx1250) — compile-verified, matrix operand order checked by reading, not run on hardware
//
#include <hip/hip_runtime.h>

#define NN    4096
#define NF    512
#define HS    128
#define HEADS 4
#define CO    64
#define NATT  9
#define LRELU 0.2f

typedef __attribute__((ext_vector_type(16))) __bf16 v16bf;
typedef __attribute__((ext_vector_type(8)))  float  v8f;
typedef __attribute__((ext_vector_type(4)))  int    v4i_;

#ifndef __has_builtin
#define __has_builtin(x) 0
#endif

#if defined(__HIP_DEVICE_COMPILE__) && __has_builtin(__builtin_amdgcn_global_load_async_to_lds_b128)
#define HAVE_ASYNC 1
#else
#define HAVE_ASYNC 0
#endif

#if HAVE_ASYNC
__device__ __forceinline__ void cp16_async(void* lds, const void* g) {
  __builtin_amdgcn_global_load_async_to_lds_b128(
      (__attribute__((address_space(1))) v4i_*)g,
      (__attribute__((address_space(3))) v4i_*)(v4i_*)lds, 0, 0);
}
#if __has_builtin(__builtin_amdgcn_s_wait_asynccnt)
#define WAIT_ASYNC() __builtin_amdgcn_s_wait_asynccnt(0)
#else
#define WAIT_ASYNC() asm volatile("s_wait_asynccnt 0x0" ::: "memory")
#endif
#else
#define WAIT_ASYNC()
#endif

__device__ __forceinline__ unsigned short f2bf(float f) {
  unsigned int u = __float_as_uint(f);
  u += 0x7FFFu + ((u >> 16) & 1u);          // round-to-nearest-even
  return (unsigned short)(u >> 16);
}
__device__ __forceinline__ float lrelu(float v){ return v > 0.f ? v : LRELU * v; }
__device__ __forceinline__ float eluf (float v){ return v > 0.f ? v : __expf(v) - 1.f; }

// WMMA-B operand swizzle: matrix [K][CCd] stored as [K/32][CCd/16][lane 0..31][elem 0..15] u16
// lane = ((r>>3)&1)*16 + (c&15);  elem = (r&7) + ((r>>4)&1)*8
__device__ __forceinline__ int swz(int r, int c, int CCd) {
  return (((r >> 5) * (CCd >> 4) + (c >> 4)) << 9)
       + ((((r >> 3) & 1) * 16 + (c & 15)) << 4)
       + (r & 7) + (((r >> 4) & 1) << 3);
}

// ---------------------------------------------------------------- small prep
__global__ void k_merge(const float* __restrict__ x, const int* __restrict__ obs,
                        const float* __restrict__ theta, float* __restrict__ xm) {
  int g = blockIdx.x * blockDim.x + threadIdx.x;
  int i = g >> 9, f = g & (NF - 1);
  float add = (obs[i] == 1) ? theta[f] : 0.f;
  xm[g] = x[g] + add;
}

__global__ void k_cvt(const float* __restrict__ s, unsigned short* __restrict__ d, int n) {
  int g = blockIdx.x * blockDim.x + threadIdx.x;
  if (g < n) d[g] = f2bf(s[g]);
}

// f32 -> bf16 with WMMA-B swizzle, batched over matrices of size K*CCd
__global__ void k_cvt_sw(const float* __restrict__ s, unsigned short* __restrict__ d,
                         int CCd, int matSize, int n) {
  int g = blockIdx.x * blockDim.x + threadIdx.x;
  if (g >= n) return;
  int m = g / matSize, rem = g - m * matSize;
  int r = rem / CCd, c = rem - r * CCd;
  d[(size_t)m * matSize + swz(r, c, CCd)] = f2bf(s[g]);
}

// va[k][which][r] = sum_c W_k[r][c] * a_k[which*256 + c]   (9 attns, d=256)
__global__ void k_wa(const float* __restrict__ W, const float* __restrict__ a,
                     const float* __restrict__ Wo, const float* __restrict__ ao,
                     float* __restrict__ va) {
  int g = blockIdx.x * blockDim.x + threadIdx.x;   // 0..9215
  int r = g & 511, which = (g >> 9) & 1, k = g >> 10;
  const float* Wk = (k < 8) ? (W + (size_t)k * 512 * 256) : Wo;
  const float* ak = (k < 8) ? (a + (size_t)k * 512 + which * 256) : (ao + which * 256);
  float s = 0.f;
  for (int c = 0; c < 256; ++c) s += Wk[(size_t)r * 256 + c] * ak[c];
  va[((size_t)k * 2 + which) * NF + r] = s;
}

// src[k][i] = xm[i,:] . va[k][0];  dst[k][i] = xm[i,:] . va[k][1]
__global__ void k_srcdst(const float* __restrict__ xm, const float* __restrict__ va,
                         float* __restrict__ src, float* __restrict__ dst) {
  int g = blockIdx.x * blockDim.x + threadIdx.x;   // 0..73727
  int i = g & (NN - 1), which = (g >> 12) & 1, k = g >> 13;
  const float* v = va + ((size_t)k * 2 + which) * NF;
  const float* xr = xm + (size_t)i * NF;
  float s = 0.f;
  for (int f = 0; f < NF; ++f) s += xr[f] * v[f];
  (which ? dst : src)[(size_t)k * NN + i] = s;
}

// per-row masked softmax statistics for all 9 attentions (one adj sweep, LDS mask)
__global__ void k_maxsum(const float* __restrict__ adj, const float* __restrict__ src,
                         const float* __restrict__ dst, float* __restrict__ mrow,
                         float* __restrict__ srow) {
  __shared__ unsigned char msk[NN];
  __shared__ float red[256];
  const int i = blockIdx.x, tid = threadIdx.x;
  const float* ar = adj + (size_t)i * NN;
  for (int j = tid; j < NN; j += 256) msk[j] = ar[j] > 0.f ? 1 : 0;
  __syncthreads();
  for (int k = 0; k < NATT; ++k) {
    const float* dk = dst + (size_t)k * NN;
    const float  si = src[(size_t)k * NN + i];
    float lm = -3.4e38f;
    for (int j = tid; j < NN; j += 256) if (msk[j]) lm = fmaxf(lm, dk[j]);
    red[tid] = lm; __syncthreads();
    for (int s = 128; s > 0; s >>= 1) { if (tid < s) red[tid] = fmaxf(red[tid], red[tid + s]); __syncthreads(); }
    const float mi = lrelu(si + red[0]);     // leaky is monotone
    __syncthreads();
    float ls = 0.f;
    for (int j = tid; j < NN; j += 256)
      if (msk[j]) ls += __expf(lrelu(si + dk[j]) - mi);
    red[tid] = ls; __syncthreads();
    for (int s = 128; s > 0; s >>= 1) { if (tid < s) red[tid] += red[tid + s]; __syncthreads(); }
    if (tid == 0) { mrow[(size_t)k * NN + i] = mi; srow[(size_t)k * NN + i] = red[0]; }
    __syncthreads();
  }
}

// ------------- bf16 WMMA GEMM: C[M,CC] = A[M,K] @ Bsw[K,CC]; C stored WMMA-B-swizzled
template<int CC, int MR>
__global__ void k_gemm(const unsigned short* __restrict__ A, const unsigned short* __restrict__ B,
                       unsigned short* __restrict__ Cm, int K,
                       size_t AzS, size_t BzS, size_t OzS) {
  constexpr int NTH = 2 * CC;
  constexpr int MT  = MR / 16;
  static_assert(MR * 4 == NTH, "A panel chunks == threads");
  __shared__ __align__(16) unsigned int   At[2][MR * 16];  // [MR][32] bf16 row-major
  __shared__ __align__(16) unsigned short Br[2][32 * CC];  // swizzled B tile
  const int tid = threadIdx.x, z = blockIdx.z;
  const unsigned short* Az = A + (size_t)z * AzS;
  const unsigned short* Bz = B + (size_t)z * BzS;
  unsigned short*       Cz = Cm + (size_t)z * OzS;
  const int i0 = blockIdx.x * MR;
  const int lane = tid & 31, w = tid >> 5, half = lane >> 4, ln = lane & 15;
  const int c0w = w * 16;
  const int am = tid >> 2, aq = tid & 3;           // A panel: 4 chunks / row
  v8f acc[MT] = {};

  auto stage = [&](int buf, int k0) {
    const unsigned short* bsrc = Bz + (size_t)(k0 >> 5) * 32 * CC;
#if HAVE_ASYNC
    cp16_async(&At[buf][am * 16 + aq * 4], Az + (size_t)(i0 + am) * K + k0 + aq * 8);
    #pragma unroll
    for (int it = 0; it < 2; ++it) {
      int ch = tid + it * NTH;
      cp16_async(&Br[buf][ch * 8], bsrc + ch * 8);
    }
#else
    uint4 ta = *(const uint4*)(Az + (size_t)(i0 + am) * K + k0 + aq * 8);
    uint4 tb0 = *(const uint4*)(bsrc + tid * 8);
    uint4 tb1 = *(const uint4*)(bsrc + (tid + NTH) * 8);
    *(uint4*)&At[buf][am * 16 + aq * 4] = ta;
    *(uint4*)&Br[buf][tid * 8] = tb0;
    *(uint4*)&Br[buf][(tid + NTH) * 8] = tb1;
#endif
  };

  const int S = K >> 5;
  stage(0, 0);
  WAIT_ASYNC();
  __syncthreads();
  for (int s = 0; s < S; ++s) {
    const int buf = s & 1;
    union { v16bf v; uint4 q[2]; } Bv;
    const uint4* Bq = (const uint4*)&Br[buf][((c0w >> 4) << 9) + (lane << 4)];
    Bv.q[0] = Bq[0]; Bv.q[1] = Bq[1];
    #pragma unroll
    for (int rt = 0; rt < MT; ++rt) {
      union { v16bf v; unsigned int u[8]; } Av;
      const int mr = rt * 16 + ln;
      #pragma unroll
      for (int p = 0; p < 4; ++p) {
        Av.u[p]     = At[buf][mr * 16 + half * 4 + p];
        Av.u[4 + p] = At[buf][mr * 16 + 8 + half * 4 + p];
      }
      acc[rt] = __builtin_amdgcn_wmma_f32_16x16x32_bf16(false, Av.v, false, Bv.v,
                                                        (short)0, acc[rt], false, false);
    }
    if (s + 1 < S) { stage(buf ^ 1, (s + 1) * 32); WAIT_ASYNC(); }
    __syncthreads();
  }
  #pragma unroll
  for (int rt = 0; rt < MT; ++rt)
    #pragma unroll
    for (int v = 0; v < 8; ++v) {
      int i = i0 + rt * 16 + half * 8 + v;
      Cz[swz(i, c0w + ln, CC)] = f2bf(acc[rt][v]);   // swizzled for the attention consumer
    }
}

// -------------------------- fused masked-softmax attention:  out = elu( P @ V / s )
// V arrives WMMA-B-swizzled; P recomputed per tile; double-buffered, 2-step adj prefetch
template<int CC, int MR, bool OUTF32>
__global__ void k_attn(const float* __restrict__ adj, const float* __restrict__ src,
                       const float* __restrict__ dst, const float* __restrict__ mrow,
                       const float* __restrict__ srow, const unsigned short* __restrict__ V,
                       void* __restrict__ outv, int kAtt0, int kStride,
                       size_t VzS, int outLd, size_t OzS) {
  constexpr int NTH = 2 * CC;
  constexpr int MT  = MR / 16;
  constexpr int PIT = (MR * 32) / (NTH * 4);
  constexpr int S   = NN / 32;
  __shared__ __align__(16) unsigned int   At[2][MR * 16];  // P tile [MR][32] bf16
  __shared__ __align__(16) unsigned short Br[2][32 * CC];  // swizzled V tile
  __shared__ float s_src[MR], s_m[MR], s_s[MR];
  const int tid = threadIdx.x, z = blockIdx.z;
  const int katt = kAtt0 + z * kStride;
  const float* srck = src  + (size_t)katt * NN;
  const float* dstk = dst  + (size_t)katt * NN;
  const float* mk   = mrow + (size_t)katt * NN;
  const float* sk   = srow + (size_t)katt * NN;
  const unsigned short* Vz = V + (size_t)z * VzS;
  const int i0 = blockIdx.x * MR;
  if (tid < MR) { s_src[tid] = srck[i0 + tid]; s_m[tid] = mk[i0 + tid]; s_s[tid] = sk[i0 + tid]; }
  const int lane = tid & 31, w = tid >> 5, half = lane >> 4, ln = lane & 15;
  const int c0w = w * 16;
  v8f acc[MT] = {};

  float4 pa[PIT], pd[PIT];
  auto loadP = [&](int jj) {               // prefetch adj/dst for tile jj into regs
    #pragma unroll
    for (int it = 0; it < PIT; ++it) {
      int e = (tid + it * NTH) * 4, m = e >> 5, kk = e & 31;
      pa[it] = *(const float4*)(adj + (size_t)(i0 + m) * NN + jj + kk);
      pd[it] = *(const float4*)(dstk + jj + kk);
    }
  };
  auto computeP = [&](int buf) {           // regs -> bf16 P tile in LDS
    #pragma unroll
    for (int it = 0; it < PIT; ++it) {
      int e = (tid + it * NTH) * 4, m = e >> 5;
      const float si = s_src[m], mi = s_m[m];
      float p0 = pa[it].x > 0.f ? __expf(lrelu(si + pd[it].x) - mi) : 0.f;
      float p1 = pa[it].y > 0.f ? __expf(lrelu(si + pd[it].y) - mi) : 0.f;
      float p2 = pa[it].z > 0.f ? __expf(lrelu(si + pd[it].z) - mi) : 0.f;
      float p3 = pa[it].w > 0.f ? __expf(lrelu(si + pd[it].w) - mi) : 0.f;
      uint2 pk;
      pk.x = (unsigned int)f2bf(p0) | ((unsigned int)f2bf(p1) << 16);
      pk.y = (unsigned int)f2bf(p2) | ((unsigned int)f2bf(p3) << 16);
      *(uint2*)&At[buf][e >> 1] = pk;
    }
  };
  auto stageV = [&](int buf, int jj) {     // linear copy of swizzled V tile
    const unsigned short* vsrc = Vz + (size_t)(jj >> 5) * 32 * CC;
#if HAVE_ASYNC
    #pragma unroll
    for (int it = 0; it < 2; ++it) {
      int ch = tid + it * NTH;
      cp16_async(&Br[buf][ch * 8], vsrc + ch * 8);
    }
#else
    uint4 t0 = *(const uint4*)(vsrc + tid * 8);
    uint4 t1 = *(const uint4*)(vsrc + (tid + NTH) * 8);
    *(uint4*)&Br[buf][tid * 8] = t0;
    *(uint4*)&Br[buf][(tid + NTH) * 8] = t1;
#endif
  };

  loadP(0);
  __syncthreads();                         // s_src/s_m visible
  computeP(0);
  stageV(0, 0);
  loadP(32);                               // prefetch step 1
  WAIT_ASYNC();
  __syncthreads();

  for (int s = 0; s < S; ++s) {
    const int buf = s & 1;
    union { v16bf v; uint4 q[2]; } Bv;
    const uint4* Bq = (const uint4*)&Br[buf][((c0w >> 4) << 9) + (lane << 4)];
    Bv.q[0] = Bq[0]; Bv.q[1] = Bq[1];
    #pragma unroll
    for (int rt = 0; rt < MT; ++rt) {
      union { v16bf v; unsigned int u[8]; } Av;
      const int mr = rt * 16 + ln;
      #pragma unroll
      for (int p = 0; p < 4; ++p) {
        Av.u[p]     = At[buf][mr * 16 + half * 4 + p];
        Av.u[4 + p] = At[buf][mr * 16 + 8 + half * 4 + p];
      }
      acc[rt] = __builtin_amdgcn_wmma_f32_16x16x32_bf16(false, Av.v, false, Bv.v,
                                                        (short)0, acc[rt], false, false);
    }
    if (s + 1 < S) {
      computeP(buf ^ 1);                   // P for step s+1 (regs prefetched)
      stageV(buf ^ 1, (s + 1) * 32);
      if (s + 2 < S) loadP((s + 2) * 32);  // adj prefetch two steps ahead
      WAIT_ASYNC();
    }
    __syncthreads();
  }

  float*          outf = (float*)outv          + (size_t)z * OzS;
  unsigned short* outh = (unsigned short*)outv + (size_t)z * OzS;
  #pragma unroll
  for (int rt = 0; rt < MT; ++rt)
    #pragma unroll
    for (int v = 0; v < 8; ++v) {
      const int mloc = rt * 16 + half * 8 + v;
      const int i = i0 + mloc, c = c0w + ln;
      const float val = eluf(acc[rt][v] / s_s[mloc]);
      if constexpr (OUTF32) outf[(size_t)i * outLd + c] = val;
      else                  outh[(size_t)i * outLd + c] = f2bf(val);
    }
}

// ---------------------------------------------------------------------------
extern "C" void kernel_launch(void* const* d_in, const int* in_sizes, int n_in,
                              void* d_out, int out_size, void* d_ws, size_t ws_size,
                              hipStream_t stream) {
  (void)in_sizes; (void)n_in; (void)out_size; (void)ws_size;
  const float* x      = (const float*)d_in[0];
  const float* adj    = (const float*)d_in[1];
  const int*   obs    = (const int*)  d_in[2];
  const float* xs     = (const float*)d_in[3];
  const float* theta  = (const float*)d_in[4];
  const float* W      = (const float*)d_in[5];
  const float* a      = (const float*)d_in[6];
  const float* Ws0    = (const float*)d_in[7];
  const float* Ws1    = (const float*)d_in[8];
  const float* Wo     = (const float*)d_in[9];
  const float* ao     = (const float*)d_in[10];
  const float* Wso    = (const float*)d_in[11];

  char* ws = (char*)d_ws;
  size_t off = 0;
  auto carve = [&](size_t bytes) { char* p = ws + off; off = (off + bytes + 255) & ~(size_t)255; return p; };
  float*          xm    = (float*)         carve((size_t)NN * NF * 4);
  unsigned short* xsb   = (unsigned short*)carve((size_t)NN * NF * 2);
  unsigned short* ws0b  = (unsigned short*)carve((size_t)HEADS * NF * HS * 2);   // swizzled
  unsigned short* ws1b  = (unsigned short*)carve((size_t)HEADS * HS * HS * 2);   // swizzled
  unsigned short* wsob  = (unsigned short*)carve((size_t)(HEADS * HS) * CO * 2); // swizzled
  float*          va    = (float*)         carve((size_t)NATT * 2 * NF * 4);
  float*          srcb  = (float*)         carve((size_t)NATT * NN * 4);
  float*          dstb  = (float*)         carve((size_t)NATT * NN * 4);
  float*          mrowb = (float*)         carve((size_t)NATT * NN * 4);
  float*          srowb = (float*)         carve((size_t)NATT * NN * 4);
  unsigned short* v0    = (unsigned short*)carve((size_t)HEADS * NN * HS * 2);   // swizzled
  unsigned short* hs0   = (unsigned short*)carve((size_t)HEADS * NN * HS * 2);   // row-major
  unsigned short* v1    = (unsigned short*)carve((size_t)HEADS * NN * HS * 2);   // swizzled
  unsigned short* hscat = (unsigned short*)carve((size_t)NN * HEADS * HS * 2);   // row-major
  unsigned short* vo    = (unsigned short*)carve((size_t)NN * CO * 2);           // swizzled

  // prep
  k_merge<<<NN * NF / 256, 256, 0, stream>>>(x, obs, theta, xm);
  k_cvt<<<NN * NF / 256, 256, 0, stream>>>(xs, xsb, NN * NF);
  k_cvt_sw<<<HEADS * NF * HS / 256, 256, 0, stream>>>(Ws0, ws0b, HS, NF * HS, HEADS * NF * HS);
  k_cvt_sw<<<HEADS * HS * HS / 256, 256, 0, stream>>>(Ws1, ws1b, HS, HS * HS, HEADS * HS * HS);
  k_cvt_sw<<<HEADS * HS * CO / 256, 256, 0, stream>>>(Wso, wsob, CO, HEADS * HS * CO, HEADS * HS * CO);
  k_wa<<<36, 256, 0, stream>>>(W, a, Wo, ao, va);
  k_srcdst<<<NATT * 2 * NN / 256, 256, 0, stream>>>(xm, va, srcb, dstb);
  k_maxsum<<<NN, 256, 0, stream>>>(adj, srcb, dstb, mrowb, srowb);

  // layer 0: V0 = x_struc @ Ws0 ; hs0 = elu(att(2h) @ V0)
  k_gemm<HS, 64><<<dim3(NN / 64, 1, HEADS), 256, 0, stream>>>(
      xsb, ws0b, v0, NF, (size_t)0, (size_t)NF * HS, (size_t)NN * HS);
  k_attn<HS, 64, false><<<dim3(NN / 64, 1, HEADS), 256, 0, stream>>>(
      adj, srcb, dstb, mrowb, srowb, v0, hs0, 0, 2, (size_t)NN * HS, HS, (size_t)NN * HS);

  // layer 1: V1 = hs0 @ Ws1 ; hscat[:, h*128:...] = elu(att(2h+1) @ V1)
  k_gemm<HS, 64><<<dim3(NN / 64, 1, HEADS), 256, 0, stream>>>(
      hs0, ws1b, v1, HS, (size_t)NN * HS, (size_t)HS * HS, (size_t)NN * HS);
  k_attn<HS, 64, false><<<dim3(NN / 64, 1, HEADS), 256, 0, stream>>>(
      adj, srcb, dstb, mrowb, srowb, v1, hscat, 1, 2, (size_t)NN * HS, HEADS * HS, (size_t)HS);

  // output: Vo = hscat @ Wso ; out = elu(att(8) @ Vo)
  k_gemm<CO, 32><<<dim3(NN / 32, 1, 1), 128, 0, stream>>>(
      hscat, wsob, vo, HEADS * HS, (size_t)0, (size_t)0, (size_t)0);
  k_attn<CO, 32, true><<<dim3(NN / 32, 1, 1), 128, 0, stream>>>(
      adj, srcb, dstb, mrowb, srowb, vo, d_out, 8, 0, (size_t)0, CO, (size_t)0);
}
